// GlobalPointer_82721070121033
// MI455X (gfx1250) — compile-verified, matrix operand order checked by reading, not run on hardware
//
#include <hip/hip_runtime.h>
#include <hip/hip_bf16.h>
#include <math.h>

// ---------------- problem constants ----------------
#define BATCH   32
#define SEQ     512
#define HIDDEN  1024
#define NHEAD   12
#define HDIM    64
#define OUTDIM  (NHEAD * HDIM * 2)   // 1536
#define INF_VAL 100000000.0f

// GEMM1 tiling
#define KC        128                // k-chunk (halves) staged per TDM transfer
#define NCHUNK    (HIDDEN / KC)      // 8
#define LDS_ROWH  (KC + 8)           // 136 halves = 272 B row stride (TDM pad 4 DW / 64 DW)

typedef __attribute__((ext_vector_type(16))) _Float16     v16h;
typedef __attribute__((ext_vector_type(8)))  float        v8f;
typedef __attribute__((ext_vector_type(4)))  unsigned int v4u;
typedef __attribute__((ext_vector_type(8)))  int          v8i;
typedef __attribute__((ext_vector_type(4)))  int          v4i;

#if defined(__has_builtin)
#if __has_builtin(__builtin_amdgcn_tensor_load_to_lds) && __has_builtin(__builtin_amdgcn_s_wait_tensorcnt)
#define USE_TDM 1
#endif
#endif
#ifndef USE_TDM
#define USE_TDM 0
#endif

// Load a 16x32 f16 WMMA operand fragment (A layout; B mirrored).
// p points at row base + k0 + (laneHi ? 8 : 0): halves 0..7 <- p[0..7],
// halves 8..15 <- p[16..23]  (ISA 7.12.2, 16-bit A-matrix 16x32).
__device__ __forceinline__ v16h load_frag_f16(const _Float16* __restrict__ p) {
    struct alignas(16) H8 { _Float16 h[8]; };
    H8 lo = *reinterpret_cast<const H8*>(p);
    H8 hi = *reinterpret_cast<const H8*>(p + 16);
    v16h v;
#pragma unroll
    for (int i = 0; i < 8; ++i) { v[i] = lo.h[i]; v[8 + i] = hi.h[i]; }
    return v;
}

#if USE_TDM
// TDM: DMA a 64-row x KC-half f16 tile of W16 (row stride HIDDEN halves) into
// LDS at lds_off. Hardware row padding: 4 DWORDs after every 64 DWORDs (256 B
// = one 128-half row) -> landed row stride = LDS_ROWH halves. ISA 8.3-8.5.
__device__ __forceinline__ void tdm_load_w_tile(const _Float16* gsrc, unsigned lds_off) {
    unsigned long long ga = (unsigned long long)(uintptr_t)gsrc;
    v4u g0;
    g0[0] = 1u;                                                // count=1, user descriptor
    g0[1] = lds_off;                                           // lds_addr (bytes)
    g0[2] = (unsigned)(ga & 0xffffffffu);                      // global_addr[31:0]
    g0[3] = (unsigned)((ga >> 32) & 0x01ffffffu) | (2u << 30); // global_addr[56:32] | type=2
    v8i g1;
    g1[0] = (int)((1u << 16)      // data_size = 2 bytes
                | (1u << 20)      // pad_enable
                | (5u << 22)      // pad_interval: 64 DWORDs (= 128 halves)
                | (3u << 25));    // pad_amount:   4 DWORDs  (= 8 halves)
    g1[1] = (int)(((unsigned)HIDDEN & 0xffffu) << 16);         // tensor_dim0[15:0]
    g1[2] = (int)((((unsigned)HIDDEN >> 16) & 0xffffu)         // tensor_dim0[31:16]
                | (((unsigned)OUTDIM & 0xffffu) << 16));       // tensor_dim1[15:0]
    g1[3] = (int)((((unsigned)OUTDIM >> 16) & 0xffffu)         // tensor_dim1[31:16]
                | ((unsigned)KC << 16));                       // tile_dim0 = 128 halves
    g1[4] = (int)64u;                                          // tile_dim1 = 64, tile_dim2 = 0
    g1[5] = (int)HIDDEN;                                       // tensor_dim0_stride (elements)
    g1[6] = 0;
    g1[7] = 0;
    v4i z4 = {0, 0, 0, 0};
#if defined(__clang_major__) && (__clang_major__ >= 23)
    v8i z8 = {0, 0, 0, 0, 0, 0, 0, 0};
    __builtin_amdgcn_tensor_load_to_lds(g0, g1, z4, z4, z8, 0);
#else
    __builtin_amdgcn_tensor_load_to_lds(g0, g1, z4, z4, 0);
#endif
}
#endif  // USE_TDM

// ============================================================
// Kernel 0: one-time fp32 -> f16 conversion (hidden and W), 8 elems/thread
// ============================================================
__global__ __launch_bounds__(256) void cvt_f16_kernel(
    const float* __restrict__ src, _Float16* __restrict__ dst, int n8)
{
    const int i = blockIdx.x * blockDim.x + threadIdx.x;
    if (i >= n8) return;
    const float* p = src + (size_t)i * 8;
    struct alignas(16) H8 { _Float16 h[8]; };
    H8 o;
#pragma unroll
    for (int j = 0; j < 8; ++j) o.h[j] = (_Float16)p[j];
    *reinterpret_cast<H8*>(dst + (size_t)i * 8) = o;
}

// ============================================================
// Kernel 1: x = hidden @ W^T + b, then RoPE, store q/k as f16.
// Block = 8 waves: 128(M) x 64(N) tile; the 64-wide N span is exactly one
// q- or k-segment of one head. W16 tile double-buffered in LDS via TDM.
// Inner loop is pure global_load_b128 / ds_load_b128 / v_wmma (no cvt).
// ============================================================
__global__ __launch_bounds__(256) void qkv_rope_kernel(
    const _Float16* __restrict__ hidden16,  // [B*S, HIDDEN] f16
    const _Float16* __restrict__ W16,       // [OUTDIM, HIDDEN] f16
    const float*    __restrict__ bias,      // [OUTDIM]
    _Float16* __restrict__ qbuf,            // [B, NHEAD, SEQ, HDIM]
    _Float16* __restrict__ kbuf)            // [B, NHEAD, SEQ, HDIM]
{
    __shared__ _Float16 wtile[2][64 * LDS_ROWH];   // 2 x 17 KB

    const int lane   = threadIdx.x & 31;
    const int wid    = threadIdx.x >> 5;             // 0..7
    const int nTile  = blockIdx.x % (OUTDIM / 64);   // 0..23
    const int mBlk   = blockIdx.x / (OUTDIM / 64);   // 0..127
    const int m0     = mBlk * 128 + wid * 16;
    const int n0     = nTile * 64;
    const int r16    = lane & 15;
    const int laneHi = lane >> 4;
    const int koff   = laneHi ? 8 : 0;

    const v8f vzero = {0.f, 0.f, 0.f, 0.f, 0.f, 0.f, 0.f, 0.f};
    v8f acc[4];
#pragma unroll
    for (int c = 0; c < 4; ++c) acc[c] = vzero;

    const _Float16* aptr = hidden16 + (size_t)(m0 + r16) * HIDDEN + koff;
    const _Float16* wseg = W16 + (size_t)n0 * HIDDEN;

#if USE_TDM
    if (wid == 0)   // one wave drives the DMA (EXEC ignored by TDM)
        tdm_load_w_tile(wseg, (unsigned)(uintptr_t)&wtile[0][0]);
#endif

    for (int ch = 0; ch < NCHUNK; ++ch) {
#if USE_TDM
        if (wid == 0) {
            if (ch + 1 < NCHUNK) {
                tdm_load_w_tile(wseg + (ch + 1) * KC,
                                (unsigned)(uintptr_t)&wtile[(ch + 1) & 1][0]);
                __builtin_amdgcn_s_wait_tensorcnt(1);   // chunk ch has landed
            } else {
                __builtin_amdgcn_s_wait_tensorcnt(0);
            }
        }
        __syncthreads();
#else
        __syncthreads();
        for (int idx = threadIdx.x; idx < 64 * KC; idx += 256) {
            const int row = idx >> 7, col = idx & (KC - 1);
            wtile[ch & 1][row * LDS_ROWH + col] = wseg[(size_t)row * HIDDEN + ch * KC + col];
        }
        __syncthreads();
#endif
        const _Float16* wbuf = &wtile[ch & 1][0];
#pragma unroll
        for (int ks = 0; ks < KC; ks += 32) {
            const v16h a = load_frag_f16(aptr + ch * KC + ks);
#pragma unroll
            for (int c = 0; c < 4; ++c) {
                const v16h bf = load_frag_f16(wbuf + (c * 16 + r16) * LDS_ROWH + ks + koff);
                acc[c] = __builtin_amdgcn_wmma_f32_16x16x32_f16(
                    false, a, false, bf, (short)0, acc[c], false, false);
            }
        }
        __syncthreads();   // protect buffer (ch&1) before it is refilled
    }

    // ---- epilogue: bias + RoPE, scatter f16 to q or k buffer ----
    const int  head = nTile >> 1;
    const bool is_q = (nTile & 1) == 0;
    _Float16* dst = is_q ? qbuf : kbuf;

    float bias_v[4];
#pragma unroll
    for (int c = 0; c < 4; ++c) bias_v[c] = bias[n0 + c * 16 + r16];

    const float lnb = logf(10000.0f);
    float invf[2];
#pragma unroll
    for (int c = 0; c < 2; ++c)
        invf[c] = expf(-lnb * (float)(c * 16 + r16) * (1.0f / 32.0f));

#pragma unroll
    for (int r = 0; r < 8; ++r) {
        const int row  = m0 + r + 8 * laneHi;   // global row in [0, B*S)
        const int bidx = row >> 9;
        const int s    = row & (SEQ - 1);
        _Float16* drow = dst + (((size_t)(bidx * NHEAD + head)) * SEQ + s) * HDIM;
#pragma unroll
        for (int c = 0; c < 2; ++c) {
            const float xlo = acc[c][r]     + bias_v[c];        // d
            const float xhi = acc[c + 2][r] + bias_v[c + 2];    // d + 32
            float sn, cs;
            __sincosf((float)s * invf[c], &sn, &cs);
            const int d = c * 16 + r16;
            drow[d]      = (_Float16)(xlo * cs - xhi * sn);
            drow[d + 32] = (_Float16)(xhi * cs + xlo * sn);
        }
    }
}

// ============================================================
// Kernel 2: scores = (q @ k^T)/8, attention mask, strict-lower tril -> -INF
// One wave per 16(M) x 64(N) tile; below-diagonal tiles skip the GEMM.
// ============================================================
__global__ __launch_bounds__(256) void attn_scores_kernel(
    const _Float16* __restrict__ qbuf,   // [B, NHEAD, SEQ, HDIM]
    const _Float16* __restrict__ kbuf,   // [B, NHEAD, SEQ, HDIM]
    const int* __restrict__ amask,       // [B, SEQ]
    float* __restrict__ out)             // [B, NHEAD, SEQ, SEQ]
{
    const int lane   = threadIdx.x & 31;
    const int wave   = blockIdx.x * (blockDim.x >> 5) + (threadIdx.x >> 5);
    const int nTile  = wave & 7;
    const int mTile  = (wave >> 3) & 31;
    const int bh     = wave >> 8;
    const int bidx   = bh / NHEAD;
    const int m0     = mTile * 16;
    const int n0     = nTile * 64;
    const int r16    = lane & 15;
    const int laneHi = lane >> 4;

    if (m0 >= n0 + 64) {      // entire tile strictly below diagonal -> -INF
#pragma unroll
        for (int r = 0; r < 8; ++r) {
            const int m = m0 + r + 8 * laneHi;
            float* p = out + ((size_t)bh * SEQ + m) * SEQ + n0;
#pragma unroll
            for (int c = 0; c < 4; ++c) p[c * 16 + r16] = -INF_VAL;
        }
        return;
    }

    const v8f vzero = {0.f, 0.f, 0.f, 0.f, 0.f, 0.f, 0.f, 0.f};
    v8f acc[4];
#pragma unroll
    for (int c = 0; c < 4; ++c) acc[c] = vzero;

    const _Float16* abase =
        qbuf + ((size_t)bh * SEQ + (m0 + r16)) * HDIM + (laneHi ? 8 : 0);
    const _Float16* bbase[4];
#pragma unroll
    for (int c = 0; c < 4; ++c)
        bbase[c] = kbuf + ((size_t)bh * SEQ + (n0 + c * 16 + r16)) * HDIM + (laneHi ? 8 : 0);

#pragma unroll
    for (int k0 = 0; k0 < HDIM; k0 += 32) {
        const v16h a = load_frag_f16(abase + k0);
#pragma unroll
        for (int c = 0; c < 4; ++c) {
            const v16h bf = load_frag_f16(bbase[c] + k0);
            acc[c] = __builtin_amdgcn_wmma_f32_16x16x32_f16(
                false, a, false, bf, (short)0, acc[c], false, false);
        }
    }

    const float scale = 0.125f;   // 1/sqrt(64)
    float aN[4];
#pragma unroll
    for (int c = 0; c < 4; ++c)
        aN[c] = (float)amask[bidx * SEQ + n0 + c * 16 + r16];

#pragma unroll
    for (int r = 0; r < 8; ++r) {
        const int m  = m0 + r + 8 * laneHi;
        const float aM = (float)amask[bidx * SEQ + m];
        float* p = out + ((size_t)bh * SEQ + m) * SEQ;
#pragma unroll
        for (int c = 0; c < 4; ++c) {
            const int ng = n0 + c * 16 + r16;
            float v;
            if (m > ng) {
                v = -INF_VAL;                         // tril(k=-1) mask
            } else {
                const float mk = aM * aN[c];
                v = acc[c][r] * scale * mk - INF_VAL * (1.0f - mk);
            }
            p[ng] = v;
        }
    }
}

// ============================================================
extern "C" void kernel_launch(void* const* d_in, const int* in_sizes, int n_in,
                              void* d_out, int out_size, void* d_ws, size_t ws_size,
                              hipStream_t stream) {
    (void)in_sizes; (void)n_in; (void)out_size; (void)ws_size;
    const float* hidden = (const float*)d_in[0];
    const int*   amask  = (const int*)d_in[1];
    const float* W      = (const float*)d_in[2];
    const float* bias   = (const float*)d_in[3];
    float*       out    = (float*)d_out;

    const size_t nHid = (size_t)BATCH * SEQ * HIDDEN;   // 16.78M
    const size_t nW   = (size_t)OUTDIM * HIDDEN;        // 1.57M
    const size_t nQK  = (size_t)BATCH * NHEAD * SEQ * HDIM;

    _Float16* hidden16 = (_Float16*)d_ws;
    _Float16* W16      = hidden16 + nHid;
    _Float16* qbuf     = W16 + nW;
    _Float16* kbuf     = qbuf + nQK;

    // Kernel 0: one-time fp32 -> f16 of hidden and W (8 elems/thread)
    const int h8 = (int)(nHid / 8);
    cvt_f16_kernel<<<(h8 + 255) / 256, 256, 0, stream>>>(hidden, hidden16, h8);
    const int w8 = (int)(nW / 8);
    cvt_f16_kernel<<<(w8 + 255) / 256, 256, 0, stream>>>(W, W16, w8);

    // Kernel 1: 128 M-blocks (128 rows each) x 24 N-tiles = 3072 blocks
    const int blocks1 = (BATCH * SEQ / 128) * (OUTDIM / 64);
    qkv_rope_kernel<<<blocks1, 256, 0, stream>>>(hidden16, W16, bias, qbuf, kbuf);

    // Kernel 2: 384 (b,h) * 32 M-tiles * 8 N-tiles = 98304 waves, 8 waves/block
    const int waves2 = (BATCH * NHEAD) * (SEQ / 16) * (SEQ / 64);
    attn_scores_kernel<<<waves2 / 8, 256, 0, stream>>>(qbuf, kbuf, amask, out);
}
